// ProbFormer_12017318494701
// MI455X (gfx1250) — matrix operand order repaired and checked by reading, not compiled, not run on hardware
//
#include <hip/hip_runtime.h>

// ---------------------------------------------------------------------------
// ProbFormer ViT forward for MI455X (gfx1250, wave32).
//  * All large GEMMs: V_WMMA_F32_16X16X32_BF16 (bf16 in, f32 accumulate),
//    double-buffered LDS tiles fed by GLOBAL_LOAD_ASYNC_TO_LDS_B128
//    (ASYNCcnt), one barrier per K-step, 4 WMMAs issued back-to-back.
//  * Weights converted to bf16 once up-front so every GEMM operand can use
//    the async-to-LDS path (no VGPR relay, no per-tile cvt).
//  * Variance path computed only for the cls token of the final layer
//    (the only place the reference consumes it).
// ---------------------------------------------------------------------------

typedef __attribute__((ext_vector_type(16))) __bf16       v16bf;
typedef __attribute__((ext_vector_type(8)))  float        v8f;
typedef __attribute__((ext_vector_type(4)))  unsigned int u4;
typedef __attribute__((ext_vector_type(4)))  float        f4;

__device__ __forceinline__ unsigned short f2bf(float x) {
  return __builtin_bit_cast(unsigned short, (__bf16)x);
}

union Pack8  { unsigned short u[8]; u4 q; };
union FragBF { v16bf v; u4 q[2]; };

#define VIT_L   4
#define VIT_D   768
#define VIT_H   12
#define VIT_DK  64
#define VIT_N   197
#define VIT_B   32
#define VIT_MLP 3072
#define VIT_NC  10
#define VIT_MT  (VIT_B * VIT_N)   // 6304 token rows
#define VIT_MP  (VIT_B * 196)     // 6272 patch rows

enum { EPI_NONE = 0, EPI_RES = 1, EPI_GELU = 2 };

#define LDP 40  // padded LDS row stride (bf16 elems): 80B rows, 16B aligned

// ---- CDNA5 async global->LDS staging (ASYNCcnt) ---------------------------
// Each thread moves 32B of A (two b128, contiguous) and 16B of W.
// INST_OFFSET applies to both the LDS and the global address (ISA 08 §4.4).
__device__ __forceinline__ void async_stage(unsigned aOff, unsigned wOff,
                                            unsigned long long gA,
                                            unsigned long long gW) {
  asm volatile(
      "global_load_async_to_lds_b128 %0, %2, off\n\t"
      "global_load_async_to_lds_b128 %0, %2, off offset:16\n\t"
      "global_load_async_to_lds_b128 %1, %3, off"
      :: "v"(aOff), "v"(wOff), "v"(gA), "v"(gW)
      : "memory");
}
__device__ __forceinline__ void async_wait0() {
  asm volatile("s_wait_asynccnt 0x0" ::: "memory");
}

// ---------------------------------------------------------------------------
// WMMA GEMM:  C[M,Nout] = A[M,K] @ W[Nout,K]^T + bias  (+ epilogue)
// Block: 256 thr (8 waves), tile 128(M) x 64(N), K-step 32, double-buffered.
// Wave w owns rows [w*16, w*16+16), all 64 cols (4 wmma accumulators).
// A, W are bf16 row-major; Nout % 64 == 0, K % 32 == 0.
// ---------------------------------------------------------------------------
template <int EPI>
__global__ __launch_bounds__(256) void gemm_bias_kernel(
    const __bf16* __restrict__ A, const __bf16* __restrict__ W,
    const float* __restrict__ bias, float* __restrict__ Cf,
    __bf16* __restrict__ Cb, const float* __restrict__ R,
    int M, int Nout, int K) {
  __shared__ __align__(16) unsigned short ldsA[2][128 * LDP];  // 2 x 10,240 B
  __shared__ __align__(16) unsigned short ldsW[2][64 * LDP];   // 2 x  5,120 B

  const int tid  = threadIdx.x;
  const int wave = tid >> 5;
  const int lane = tid & 31;
  const int m0   = blockIdx.x * 128;
  const int n0   = blockIdx.y * 64;

  v8f acc[4] = {};

  // staging assignment: A tile 128x32 -> 16 elems/thread; W tile 64x32 -> 8
  const int ar = tid >> 1;             // 0..127
  const int ac = (tid & 1) * 16;       // 0,16
  const int wr = tid >> 2;             // 0..63
  const int wc = (tid & 3) * 8;        // 0,8,16,24
  int arow = m0 + ar; if (arow > M - 1) arow = M - 1;  // clamp (reads only)
  const int wrow = n0 + wr;

  // LDS byte offsets for the async destinations (low 32 bits of flat address
  // of a __shared__ object == workgroup-relative LDS byte offset).
  const unsigned aOff[2] = {
      (unsigned)(unsigned long long)(void*)&ldsA[0][ar * LDP + ac],
      (unsigned)(unsigned long long)(void*)&ldsA[1][ar * LDP + ac]};
  const unsigned wOff[2] = {
      (unsigned)(unsigned long long)(void*)&ldsW[0][wr * LDP + wc],
      (unsigned)(unsigned long long)(void*)&ldsW[1][wr * LDP + wc]};
  const unsigned long long gA = (unsigned long long)(A + (size_t)arow * K + ac);
  const unsigned long long gW = (unsigned long long)(W + (size_t)wrow * K + wc);

  // fragment LDS bases (contiguous k = khalf*16 + i for A and B, wave32)
  const int l16   = lane & 15;
  const int khalf = lane >> 4;
  const unsigned short* paB[2] = {
      &ldsA[0][(wave * 16 + l16) * LDP + khalf * 16],
      &ldsA[1][(wave * 16 + l16) * LDP + khalf * 16]};
  const unsigned short* pwB[2] = {
      &ldsW[0][l16 * LDP + khalf * 16],
      &ldsW[1][l16 * LDP + khalf * 16]};

  const int nk = K >> 5;
  async_stage(aOff[0], wOff[0], gA, gW);
  async_wait0();
  __syncthreads();

  for (int i = 0; i < nk; ++i) {
    const int buf = i & 1;
    if (i + 1 < nk)  // prefetch next K-step into the other buffer
      async_stage(aOff[buf ^ 1], wOff[buf ^ 1],
                  gA + ((unsigned long long)(i + 1) << 6),
                  gW + ((unsigned long long)(i + 1) << 6));

    // A fragment: lane holds row m=l16; elems i = K kb+i,
    // kb = khalf*16.
    const unsigned short* pa = paB[buf];
    FragBF fa;
    fa.q[0] = *(const u4*)(pa);
    fa.q[1] = *(const u4*)(pa + 8);
    // B fragments: lane holds col n=l16; elems i = K khalf*16 + i.
    const unsigned short* pw = pwB[buf];
    FragBF fb0, fb1, fb2, fb3;
    fb0.q[0] = *(const u4*)(pw);                fb0.q[1] = *(const u4*)(pw + 8);
    fb1.q[0] = *(const u4*)(pw + 16 * LDP);     fb1.q[1] = *(const u4*)(pw + 16 * LDP + 8);
    fb2.q[0] = *(const u4*)(pw + 32 * LDP);     fb2.q[1] = *(const u4*)(pw + 32 * LDP + 8);
    fb3.q[0] = *(const u4*)(pw + 48 * LDP);     fb3.q[1] = *(const u4*)(pw + 48 * LDP + 8);

    acc[0] = __builtin_amdgcn_wmma_f32_16x16x32_bf16(false, fa.v, false, fb0.v,
                                                     (short)0, acc[0], false, false);
    acc[1] = __builtin_amdgcn_wmma_f32_16x16x32_bf16(false, fa.v, false, fb1.v,
                                                     (short)0, acc[1], false, false);
    acc[2] = __builtin_amdgcn_wmma_f32_16x16x32_bf16(false, fa.v, false, fb2.v,
                                                     (short)0, acc[2], false, false);
    acc[3] = __builtin_amdgcn_wmma_f32_16x16x32_bf16(false, fa.v, false, fb3.v,
                                                     (short)0, acc[3], false, false);

    if (i + 1 < nk) async_wait0();  // next tile landed (issued before compute)
    __syncthreads();
  }

  // ---- epilogue: C layout VGPR j -> M = j + (lane>=16 ? 8 : 0), N = l16 ----
  const int  mbase = m0 + wave * 16 + (khalf ? 8 : 0);
  const bool wf = (Cf != nullptr), wb = (Cb != nullptr);
#pragma unroll
  for (int t = 0; t < 4; ++t) {
    const int   n  = n0 + t * 16 + l16;
    const float bn = bias[n];
#pragma unroll
    for (int j = 0; j < 8; ++j) {
      const int m = mbase + j;
      if (m < M) {
        const size_t off = (size_t)m * Nout + n;
        float c = acc[t][j] + bn;
        if constexpr (EPI == EPI_RES)  c += R[off];
        if constexpr (EPI == EPI_GELU) c = 0.5f * c * (1.0f + erff(c * 0.70710678118f));
        if (wf) Cf[off] = c;
        if (wb) Cb[off] = (__bf16)c;
      }
    }
  }
}

// ---------------------------------------------------------------------------
// f32 -> bf16 bulk conversion (weights; n % 2048 == 0)
// ---------------------------------------------------------------------------
__global__ __launch_bounds__(256) void convert_bf16_kernel(
    const float* __restrict__ src, __bf16* __restrict__ dst) {
  const size_t i = ((size_t)blockIdx.x * 256 + threadIdx.x) * 8;
  f4 a = *(const f4*)(src + i), b = *(const f4*)(src + i + 4);
  Pack8 p;
  p.u[0] = f2bf(a.x); p.u[1] = f2bf(a.y); p.u[2] = f2bf(a.z); p.u[3] = f2bf(a.w);
  p.u[4] = f2bf(b.x); p.u[5] = f2bf(b.y); p.u[6] = f2bf(b.z); p.u[7] = f2bf(b.w);
  *(u4*)((unsigned short*)dst + i) = p.q;
}

// ---------------------------------------------------------------------------
// LayerNorm over rows of length 768; OUT is f32 or bf16.
// ---------------------------------------------------------------------------
template <typename OUT>
__global__ __launch_bounds__(256) void ln_kernel(
    const float* __restrict__ x, size_t rowStride, const float* __restrict__ s,
    const float* __restrict__ b, OUT* __restrict__ out, size_t outStride) {
  __shared__ float r1[256];
  __shared__ float r2[256];
  const int tid = threadIdx.x;
  const float* xr = x + (size_t)blockIdx.x * rowStride;
  const float v0 = xr[tid], v1 = xr[tid + 256], v2 = xr[tid + 512];
  r1[tid] = v0 + v1 + v2;
  r2[tid] = v0 * v0 + v1 * v1 + v2 * v2;
  __syncthreads();
  for (int off = 128; off; off >>= 1) {
    if (tid < off) { r1[tid] += r1[tid + off]; r2[tid] += r2[tid + off]; }
    __syncthreads();
  }
  const float mean = r1[0] * (1.0f / VIT_D);
  const float var  = r2[0] * (1.0f / VIT_D) - mean * mean;
  const float rs   = rsqrtf(var + 1e-6f);
  OUT* o = out + (size_t)blockIdx.x * outStride;
  o[tid]       = (OUT)((v0 - mean) * rs * s[tid]       + b[tid]);
  o[tid + 256] = (OUT)((v1 - mean) * rs * s[tid + 256] + b[tid + 256]);
  o[tid + 512] = (OUT)((v2 - mean) * rs * s[tid + 512] + b[tid + 512]);
}

// ---------------------------------------------------------------------------
// Attention (means path): one wave per output row (b,h,n). 8 rows / block.
// Writes z_means in f32 (cls-token consumers) and bf16 (o-proj GEMM).
// ---------------------------------------------------------------------------
__global__ __launch_bounds__(256) void attn_kernel(
    const float* __restrict__ qkv, float* __restrict__ zm,
    __bf16* __restrict__ zmb) {
  __shared__ float qs[8][VIT_DK];
  __shared__ float ps[8][200];
  const int b = blockIdx.x / VIT_H, h = blockIdx.x % VIT_H;
  const int w = threadIdx.x >> 5, lane = threadIdx.x & 31;
  const int n = blockIdx.y * 8 + w;
  const bool valid = (n < VIT_N);
  const int nc = valid ? n : VIT_N - 1;

  const float* qr = qkv + ((size_t)(b * VIT_N + nc)) * (3 * VIT_D) + h * VIT_DK;
  qs[w][lane * 2]     = qr[lane * 2];
  qs[w][lane * 2 + 1] = qr[lane * 2 + 1];
  __syncthreads();

  for (int m = lane; m < VIT_N; m += 32) {
    const float* kr = qkv + ((size_t)(b * VIT_N + m)) * (3 * VIT_D) + VIT_D + h * VIT_DK;
    float acc = 0.0f;
#pragma unroll
    for (int d = 0; d < VIT_DK; d += 4) {
      f4 kk = *(const f4*)(kr + d);
      acc += qs[w][d] * kk.x + qs[w][d + 1] * kk.y + qs[w][d + 2] * kk.z + qs[w][d + 3] * kk.w;
    }
    ps[w][m] = acc * 0.125f;  // 1/sqrt(64)
  }
  __syncthreads();

  float mx = -1e30f;
  for (int m = lane; m < VIT_N; m += 32) mx = fmaxf(mx, ps[w][m]);
  for (int off = 16; off; off >>= 1) mx = fmaxf(mx, __shfl_xor(mx, off, 32));
  float sum = 0.0f;
  for (int m = lane; m < VIT_N; m += 32) {
    float e = __expf(ps[w][m] - mx);
    ps[w][m] = e;
    sum += e;
  }
  for (int off = 16; off; off >>= 1) sum += __shfl_xor(sum, off, 32);
  const float inv = 1.0f / sum;
  __syncthreads();

  const int d0 = lane * 2;
  float a0 = 0.0f, a1 = 0.0f;
  for (int m = 0; m < VIT_N; ++m) {
    const float p = ps[w][m];
    const float* vr = qkv + ((size_t)(b * VIT_N + m)) * (3 * VIT_D) + 2 * VIT_D + h * VIT_DK + d0;
    a0 += p * vr[0];
    a1 += p * vr[1];
  }
  if (valid) {
    const size_t o = ((size_t)(b * VIT_N + n)) * VIT_D + h * VIT_DK + d0;
    const float r0 = a0 * inv, r1v = a1 * inv;
    zm[o]      = r0;
    zm[o + 1]  = r1v;
    zmb[o]     = (__bf16)r0;
    zmb[o + 1] = (__bf16)r1v;
  }
}

// ---------------------------------------------------------------------------
// Final-layer cls-token variance: z_vars[b,h*64+d] = sum_m p0[m]*((v-hm0)^2+e^s)
// ---------------------------------------------------------------------------
__global__ __launch_bounds__(32) void cls_var_kernel(
    const float* __restrict__ qkv, const float* __restrict__ zm,
    const float* __restrict__ sigmas, int layer, float* __restrict__ zv) {
  __shared__ float qs[VIT_DK];
  __shared__ float ps[200];
  const int b = blockIdx.x / VIT_H, h = blockIdx.x % VIT_H;
  const int lane = threadIdx.x;

  const float* qr = qkv + ((size_t)b * VIT_N) * (3 * VIT_D) + h * VIT_DK;
  qs[lane * 2]     = qr[lane * 2];
  qs[lane * 2 + 1] = qr[lane * 2 + 1];
  __syncthreads();

  for (int m = lane; m < VIT_N; m += 32) {
    const float* kr = qkv + ((size_t)(b * VIT_N + m)) * (3 * VIT_D) + VIT_D + h * VIT_DK;
    float acc = 0.0f;
#pragma unroll
    for (int d = 0; d < VIT_DK; d += 4) {
      f4 kk = *(const f4*)(kr + d);
      acc += qs[d] * kk.x + qs[d + 1] * kk.y + qs[d + 2] * kk.z + qs[d + 3] * kk.w;
    }
    ps[m] = acc * 0.125f;
  }
  __syncthreads();
  float mx = -1e30f;
  for (int m = lane; m < VIT_N; m += 32) mx = fmaxf(mx, ps[m]);
  for (int off = 16; off; off >>= 1) mx = fmaxf(mx, __shfl_xor(mx, off, 32));
  float sum = 0.0f;
  for (int m = lane; m < VIT_N; m += 32) {
    float e = __expf(ps[m] - mx);
    ps[m] = e;
    sum += e;
  }
  for (int off = 16; off; off >>= 1) sum += __shfl_xor(sum, off, 32);
  const float inv = 1.0f / sum;
  __syncthreads();

  const int d0 = lane * 2;
  const size_t hm = ((size_t)b * VIT_N) * VIT_D + h * VIT_DK + d0;
  const float hm0 = zm[hm], hm1 = zm[hm + 1];
  const float es = __expf(sigmas[layer]);
  float a0 = 0.0f, a1 = 0.0f;
  for (int m = 0; m < VIT_N; ++m) {
    const float p = ps[m] * inv;
    const float* vr = qkv + ((size_t)(b * VIT_N + m)) * (3 * VIT_D) + 2 * VIT_D + h * VIT_DK + d0;
    const float dv0 = vr[0] - hm0, dv1 = vr[1] - hm1;
    a0 += p * (dv0 * dv0 + es);
    a1 += p * (dv1 * dv1 + es);
  }
  zv[(size_t)b * VIT_D + h * VIT_DK + d0]     = a0;
  zv[(size_t)b * VIT_D + h * VIT_DK + d0 + 1] = a1;
}

// ---------------------------------------------------------------------------
// Cls-token projections: ym = z_means[:,0] @ w_o^T + b_o ; yv = z_vars @ (w_o^2)^T
// ---------------------------------------------------------------------------
__global__ __launch_bounds__(256) void cls_proj_kernel(
    const float* __restrict__ zm, const float* __restrict__ zv,
    const float* __restrict__ wo, const float* __restrict__ bo,
    float* __restrict__ ym, float* __restrict__ yv) {
  const int b = blockIdx.x;
  const float* zr = zm + ((size_t)b * VIT_N) * VIT_D;  // token 0
  const float* vr = zv + (size_t)b * VIT_D;
  for (int e = 0; e < 3; ++e) {
    const int j = threadIdx.x + e * 256;
    const float* wr = wo + (size_t)j * VIT_D;
    float sm = 0.0f, sv = 0.0f;
    for (int d = 0; d < VIT_D; d += 4) {
      f4 w4 = *(const f4*)(wr + d);
      f4 z4 = *(const f4*)(zr + d);
      f4 v4 = *(const f4*)(vr + d);
      sm += z4.x * w4.x + z4.y * w4.y + z4.z * w4.z + z4.w * w4.w;
      sv += v4.x * w4.x * w4.x + v4.y * w4.y * w4.y + v4.z * w4.z * w4.z + v4.w * w4.w * w4.w;
    }
    ym[(size_t)b * VIT_D + j] = sm + bo[j];
    yv[(size_t)b * VIT_D + j] = sv;
  }
}

// ---------------------------------------------------------------------------
// im2row for 16x16 patches -> bf16
// ---------------------------------------------------------------------------
__global__ __launch_bounds__(256) void patch_kernel(
    const float* __restrict__ in, __bf16* __restrict__ pbf) {
  const int pm = blockIdx.x;  // b*196 + p
  const int b = pm / 196, p = pm % 196;
  const int gy = p / 14, gx = p % 14;
  for (int e = 0; e < 3; ++e) {
    const int k = threadIdx.x + e * 256;
    const int c = k >> 8, r = k & 255, py = r >> 4, px = r & 15;
    const size_t src = (((size_t)b * 3 + c) * 224 + (gy * 16 + py)) * 224 + gx * 16 + px;
    pbf[(size_t)pm * VIT_D + k] = (__bf16)in[src];
  }
}

// x[b,0]=class_token+pos[0]; x[b,1+p]=patch_embed[b,p]+pos[1+p]
__global__ __launch_bounds__(256) void assemble_kernel(
    const float* __restrict__ pe, const float* __restrict__ ct,
    const float* __restrict__ pos, float* __restrict__ x) {
  const int bn = blockIdx.x;  // b*197 + n
  const int b = bn / VIT_N, n = bn % VIT_N;
  for (int e = 0; e < 3; ++e) {
    const int d = threadIdx.x + e * 256;
    const float v = (n == 0) ? ct[d] : pe[((size_t)b * 196 + (n - 1)) * VIT_D + d];
    x[(size_t)bn * VIT_D + d] = v + pos[(size_t)n * VIT_D + d];
  }
}

// logits / logits_mean / sqrt(logits_var)  ->  d_out[0:320 | 320:640 | 640:960]
__global__ __launch_bounds__(32) void final_kernel(
    const float* __restrict__ xf, const float* __restrict__ ym,
    const float* __restrict__ yv, const float* __restrict__ cw,
    const float* __restrict__ cb, float* __restrict__ out) {
  const int b = blockIdx.x, t = threadIdx.x;
  if (t >= 30) return;
  const int kind = t / VIT_NC, c = t % VIT_NC;
  const float* wr = cw + (size_t)c * VIT_D;
  const float* src = (kind == 0) ? xf + (size_t)b * VIT_D
                   : (kind == 1) ? ym + (size_t)b * VIT_D
                                 : yv + (size_t)b * VIT_D;
  float acc = 0.0f;
  for (int d = 0; d < VIT_D; ++d) {
    const float w = wr[d];
    acc += (kind == 2) ? src[d] * w * w : src[d] * w;
  }
  if (kind == 0)      out[b * VIT_NC + c]       = acc + cb[c];
  else if (kind == 1) out[320 + b * VIT_NC + c] = acc + cb[c];
  else                out[640 + b * VIT_NC + c] = sqrtf(acc);
}

// ---------------------------------------------------------------------------
// Driver
// ---------------------------------------------------------------------------
extern "C" void kernel_launch(void* const* d_in, const int* in_sizes, int n_in,
                              void* d_out, int out_size, void* d_ws, size_t ws_size,
                              hipStream_t stream) {
  (void)in_sizes; (void)n_in; (void)out_size; (void)ws_size;
  const float* input   = (const float*)d_in[0];
  const float* conv_w  = (const float*)d_in[1];
  const float* conv_b  = (const float*)d_in[2];
  const float* cls_tok = (const float*)d_in[3];
  const float* pos     = (const float*)d_in[4];
  const float* ln1_s   = (const float*)d_in[5];
  const float* ln1_b   = (const float*)d_in[6];
  const float* w_qkv   = (const float*)d_in[7];
  const float* b_qkv   = (const float*)d_in[8];
  const float* w_o     = (const float*)d_in[9];
  const float* b_o     = (const float*)d_in[10];
  const float* sigmas  = (const float*)d_in[11];
  const float* ln2_s   = (const float*)d_in[12];
  const float* ln2_b   = (const float*)d_in[13];
  const float* w1      = (const float*)d_in[14];
  const float* b1      = (const float*)d_in[15];
  const float* w2      = (const float*)d_in[16];
  const float* b2      = (const float*)d_in[17];
  const float* lnf_s   = (const float*)d_in[18];
  const float* lnf_b   = (const float*)d_in[19];
  const float* cls_w   = (const float*)d_in[20];
  const float* cls_b   = (const float*)d_in[21];

  // ---- workspace carve-out (cursor allocator, 256B aligned) ----
  char* cur = (char*)d_ws;
  auto alloc = [&](size_t sz) { char* r = cur; cur += (sz + 255) & ~(size_t)255; return r; };
  float*  X    = (float*) alloc((size_t)VIT_MT * VIT_D * 4);
  __bf16* HBF  = (__bf16*)alloc((size_t)VIT_MT * VIT_D * 2);       // LN out / patches
  float*  BIG  = (float*) alloc((size_t)VIT_MT * 3 * VIT_D * 4);   // qkv f32
  __bf16* MBF  = (__bf16*)BIG;                                     // MLP act bf16 (alias)
  float*  ZM   = (float*) alloc((size_t)VIT_MT * VIT_D * 4);
  __bf16* ZMB  = (__bf16*)alloc((size_t)VIT_MT * VIT_D * 2);
  float*  ZVC  = (float*) alloc((size_t)VIT_B * VIT_D * 4);
  float*  YMC  = (float*) alloc((size_t)VIT_B * VIT_D * 4);
  float*  YVC  = (float*) alloc((size_t)VIT_B * VIT_D * 4);
  float*  XFC  = (float*) alloc((size_t)VIT_B * VIT_D * 4);
  __bf16* WBC  = (__bf16*)alloc((size_t)VIT_D * VIT_D * 2);                    // conv
  __bf16* WBQ  = (__bf16*)alloc((size_t)VIT_L * 3 * VIT_D * VIT_D * 2);        // qkv
  __bf16* WBO  = (__bf16*)alloc((size_t)VIT_L * VIT_D * VIT_D * 2);            // o
  __bf16* WB1  = (__bf16*)alloc((size_t)VIT_L * VIT_MLP * VIT_D * 2);          // mlp1
  __bf16* WB2  = (__bf16*)alloc((size_t)VIT_L * VIT_D * VIT_MLP * 2);          // mlp2
  float*  out  = (float*)d_out;

  // ---- one-time weight conversion to bf16 (all counts % 2048 == 0) ----
  convert_bf16_kernel<<<(VIT_D * VIT_D) / 2048, 256, 0, stream>>>(conv_w, WBC);
  convert_bf16_kernel<<<(VIT_L * 3 * VIT_D * VIT_D) / 2048, 256, 0, stream>>>(w_qkv, WBQ);
  convert_bf16_kernel<<<(VIT_L * VIT_D * VIT_D) / 2048, 256, 0, stream>>>(w_o, WBO);
  convert_bf16_kernel<<<(VIT_L * VIT_MLP * VIT_D) / 2048, 256, 0, stream>>>(w1, WB1);
  convert_bf16_kernel<<<(VIT_L * VIT_D * VIT_MLP) / 2048, 256, 0, stream>>>(w2, WB2);

  // ---- patch embedding ----
  patch_kernel<<<VIT_MP, 256, 0, stream>>>(input, HBF);
  gemm_bias_kernel<EPI_NONE><<<dim3(49, 12), 256, 0, stream>>>(
      HBF, WBC, conv_b, BIG, nullptr, nullptr, VIT_MP, VIT_D, VIT_D);
  assemble_kernel<<<VIT_B * VIT_N, 256, 0, stream>>>(BIG, cls_tok, pos, X);

  // ---- transformer layers ----
  for (int l = 0; l < VIT_L; ++l) {
    ln_kernel<__bf16><<<VIT_MT, 256, 0, stream>>>(
        X, VIT_D, ln1_s + l * VIT_D, ln1_b + l * VIT_D, HBF, VIT_D);
    gemm_bias_kernel<EPI_NONE><<<dim3(50, 36), 256, 0, stream>>>(
        HBF, WBQ + (size_t)l * 3 * VIT_D * VIT_D, b_qkv + l * 3 * VIT_D,
        BIG, nullptr, nullptr, VIT_MT, 3 * VIT_D, VIT_D);
    attn_kernel<<<dim3(VIT_B * VIT_H, 25), 256, 0, stream>>>(BIG, ZM, ZMB);
    if (l == VIT_L - 1) {
      cls_var_kernel<<<VIT_B * VIT_H, 32, 0, stream>>>(BIG, ZM, sigmas, l, ZVC);
      cls_proj_kernel<<<VIT_B, 256, 0, stream>>>(
          ZM, ZVC, w_o + (size_t)l * VIT_D * VIT_D, b_o + l * VIT_D, YMC, YVC);
    }
    gemm_bias_kernel<EPI_RES><<<dim3(50, 12), 256, 0, stream>>>(
        ZMB, WBO + (size_t)l * VIT_D * VIT_D, b_o + l * VIT_D,
        X, nullptr, X, VIT_MT, VIT_D, VIT_D);  // in-place residual
    ln_kernel<__bf16><<<VIT_MT, 256, 0, stream>>>(
        X, VIT_D, ln2_s + l * VIT_D, ln2_b + l * VIT_D, HBF, VIT_D);
    gemm_bias_kernel<EPI_GELU><<<dim3(50, 48), 256, 0, stream>>>(
        HBF, WB1 + (size_t)l * VIT_MLP * VIT_D, b1 + l * VIT_MLP,
        nullptr, MBF, nullptr, VIT_MT, VIT_MLP, VIT_D);
    gemm_bias_kernel<EPI_RES><<<dim3(50, 12), 256, 0, stream>>>(
        MBF, WB2 + (size_t)l * VIT_D * VIT_MLP, b2 + l * VIT_D,
        X, nullptr, X, VIT_MT, VIT_D, VIT_MLP);
  }

  // ---- head ----
  ln_kernel<float><<<VIT_B, 256, 0, stream>>>(
      X, (size_t)VIT_N * VIT_D, lnf_s, lnf_b, XFC, VIT_D);
  final_kernel<<<VIT_B, 32, 0, stream>>>(XFC, YMC, YVC, cls_w, cls_b, out);
}
